// EventToDepthModel_88210038325685
// MI455X (gfx1250) — matrix operand order repaired and checked by reading, not compiled, not run on hardware
//
#include <hip/hip_runtime.h>
#include <cstdint>
#include <cstddef>

typedef __attribute__((ext_vector_type(16))) __bf16 v16bf;
typedef __attribute__((ext_vector_type(8)))  float  v8f;

#define BB 8
#define NN 65536
#define HH 128
#define WW 128
#define DD 128

__device__ __forceinline__ float sigf(float x) { return 1.0f / (1.0f + __expf(-x)); }

// ---------------------------------------------------------------- zero
__global__ void zero_f32_kernel(float* __restrict__ p, int n) {
  int i = blockIdx.x * blockDim.x + threadIdx.x;
  if (i < n) p[i] = 0.0f;
}

// ---------------------------------------------------------------- embed + masked scatter-add
__global__ void embed_scatter_kernel(const float* __restrict__ ev,
                                     const unsigned char* __restrict__ mask,
                                     const float* __restrict__ ew,     // [4,128]
                                     const float* __restrict__ ebias,  // [128]
                                     float* __restrict__ grid,         // [B,H,W,128] f32
                                     float* __restrict__ counts) {     // [B,H,W]
  const int e = blockIdx.x;                 // B*N events
  if (mask[e] == 0) return;
  const float* ep = ev + (size_t)e * 4;
  const float e0 = ep[0], e1 = ep[1], e2 = ep[2], e3 = ep[3];
  int x = (int)(e1 * (float)WW); x = x < 0 ? 0 : (x > WW - 1 ? WW - 1 : x);
  int y = (int)(e2 * (float)HH); y = y < 0 ? 0 : (y > HH - 1 ? HH - 1 : y);
  const int b = e >> 16;                    // N = 65536
  const int d = threadIdx.x;
  const float f = ebias[d] + e0 * ew[d] + e1 * ew[DD + d] + e2 * ew[2 * DD + d] + e3 * ew[3 * DD + d];
  const size_t pix = (size_t)(b * HH + y) * WW + x;
  atomicAdd(grid + pix * DD + d, f);
  if (d == 0) atomicAdd(counts + pix, 1.0f);
}

// ---------------------------------------------------------------- normalize + convert to bf16
__global__ void normalize_kernel(const float* __restrict__ grid,
                                 const float* __restrict__ counts,
                                 __bf16* __restrict__ out, int total) {
  int i = blockIdx.x * blockDim.x + threadIdx.x;
  if (i >= total) return;
  const float c = counts[i >> 7];           // D = 128 channels innermost
  out[i] = (__bf16)(grid[i] / (c + 1e-6f));
}

// ---------------------------------------------------------------- weight pre-pack into A-fragment order
// Packed layout: wp[mTile][kStep][lane][e]  (contiguous 16 bf16 per lane)
// A-matrix 16x32 bf16 layout: m = lane&15; half = lane>>4;
//   e<8  -> k_local = half*8 + e ; e>=8 -> k_local = 16 + half*8 + (e-8)
template<int CINFULL, int CIN, int COUT>
__global__ void pack_weights_kernel(const float* __restrict__ w, __bf16* __restrict__ wp, int total) {
  constexpr int KSTEPS = CIN * 9 / 32;
  int idx = blockIdx.x * blockDim.x + threadIdx.x;
  if (idx >= total) return;
  const int e    = idx & 15;
  const int lane = (idx >> 4) & 31;
  const int ks   = (idx >> 9) % KSTEPS;
  const int mt   = idx / (KSTEPS * 512);
  const int half = lane >> 4;
  const int ml   = lane & 15;
  const int kl   = (e < 8) ? (half * 8 + e) : (16 + half * 8 + (e - 8));
  const int k    = ks * 32 + kl;
  const int c    = k & (CIN - 1);
  const int tap  = k / CIN;
  const int ky   = tap / 3, kx = tap - ky * 3;
  const int m    = mt * 16 + ml;
  wp[idx] = (__bf16)w[(((size_t)m * CINFULL + c) * 3 + ky) * 3 + kx];
}

// ---------------------------------------------------------------- implicit-GEMM 3x3 conv via WMMA
// Block = one (NT*16)-pixel row strip; TILES waves, each wave = 1 M-tile x NT N-tiles
// (each A fragment from L2 feeds NT WMMAs). LSTM mode computes only i/o/g gate
// tiles (forget gate is dead because c0 == 0): wave wv -> M-tile (wv<4 ? wv : wv+4).
// Patch staged in LDS in B-fragment order: [kStep][nt][half][n16][e16].
// f32 output tile buffer aliases the patch buffer (patch dead after main loop).
template<int CIN, int TILES, int NT, bool LSTM>
__global__ __launch_bounds__(TILES * 32)
void conv3x3_wmma_kernel(const __bf16* __restrict__ in,     // [B,H,W,CIN] bf16
                         const __bf16* __restrict__ wpack,  // packed A fragments
                         const float*  __restrict__ bias,   // f32 bias (full channel count)
                         __bf16* __restrict__ out) {        // [B,H,W,COUTeff] bf16
  constexpr int K = CIN * 9;
  constexpr int KSTEPS = K / 32;
  constexpr int NPX = NT * 16;                    // pixels per block
  constexpr int NTHREADS = TILES * 32;
  constexpr int PATCH_BYTES = K * NPX * 2;        // bf16 patch
  constexpr int OUT_BYTES   = TILES * 16 * NPX * 4;
  constexpr int SMEM_BYTES  = PATCH_BYTES > OUT_BYTES ? PATCH_BYTES : OUT_BYTES;
  __shared__ __align__(32) char sMem[SMEM_BYTES];
  __bf16* sPatch = (__bf16*)sMem;                 // phase 1
  float*  sOut   = (float*)sMem;                  // phase 2 (aliased)

  const int tid = threadIdx.x;
  const int bid = blockIdx.x;
  constexpr int SPR = WW / NPX;                   // strips per row
  const int x0 = (bid % SPR) * NPX;
  const int y  = (bid / SPR) % HH;
  const int b  = bid / (SPR * HH);

  // ---- stage zero-padded patch for NPX pixels into LDS in B-fragment order
  for (int idx = tid; idx < K * NPX; idx += NTHREADS) {
    const int np  = idx & (NPX - 1);
    const int k   = idx / NPX;
    const int c   = k & (CIN - 1);
    const int tap = k / CIN;
    const int ky  = tap / 3, kx = tap - ky * 3;
    const int yy  = y + ky - 1;
    const int xx  = x0 + np + kx - 1;
    __bf16 v = (__bf16)0.0f;
    if (yy >= 0 && yy < HH && xx >= 0 && xx < WW)
      v = in[(size_t)((b * HH + yy) * WW + xx) * CIN + c];
    const int ks   = k >> 5;
    const int half = (k >> 3) & 1;
    const int ee   = (k & 7) + (((k >> 4) & 1) << 3);
    const int nt   = np >> 4;
    const int n    = np & 15;
    sPatch[(((((ks * NT + nt) * 2 + half) * 16) + n) << 4) + ee] = v;
  }
  __syncthreads();

  // ---- WMMA main loop: 1 A fragment -> NT WMMAs
  const int lane  = tid & 31;
  const int wv    = tid >> 5;                               // wave index
  const int mt    = LSTM ? (wv < 4 ? wv : wv + 4) : wv;     // M-tile (skip f gate)
  const int halfL = lane >> 4;
  const int nL    = lane & 15;

  v8f acc[NT];
#pragma unroll
  for (int nt = 0; nt < NT; ++nt) acc[nt] = (v8f){};

  const v16bf* aFrag = reinterpret_cast<const v16bf*>(wpack) + (size_t)mt * KSTEPS * 32 + lane;
  const v16bf* bBase = reinterpret_cast<const v16bf*>(&sPatch[0]);
  for (int ks = 0; ks < KSTEPS; ++ks) {
    const v16bf a = aFrag[ks * 32];
#pragma unroll
    for (int nt = 0; nt < NT; ++nt) {
      const v16bf bf = bBase[((ks * NT + nt) * 2 + halfL) * 16 + nL];
      acc[nt] = __builtin_amdgcn_wmma_f32_16x16x32_bf16(false, a, false, bf,
                                                        (short)0, acc[nt], false, false);
    }
  }
  __syncthreads();   // patch reads done; safe to alias as sOut

  // ---- D tiles (+bias) to LDS (compact rows: wave wv at rows wv*16..):
#pragma unroll
  for (int r = 0; r < 8; ++r) {
    const int rowL = wv * 16 + halfL * 8 + r;      // compact row in sOut
    const int m    = mt * 16 + halfL * 8 + r;      // real out channel (for bias)
    const float bv = bias[m];
#pragma unroll
    for (int nt = 0; nt < NT; ++nt)
      sOut[rowL * NPX + nt * 16 + nL] = acc[nt][r] + bv;
  }
  __syncthreads();

  // ---- epilogue over NPX pixels
  const size_t pixBase = (size_t)((b * HH + y) * WW + x0);
  if (LSTM) {
    // compact rows: i -> 0..63, o -> 64..127, g -> 128..191 ; c0=0 => c=sig(i)*tanh(g)
    for (int idx = tid; idx < 64 * NPX; idx += NTHREADS) {
      const int ch = idx & 63;
      const int n  = idx >> 6;
      const float gi = sOut[(ch)       * NPX + n];
      const float go = sOut[(64 + ch)  * NPX + n];
      const float gg = sOut[(128 + ch) * NPX + n];
      const float cc = sigf(gi) * tanhf(gg);
      const float hh = sigf(go) * tanhf(cc);
      out[(pixBase + n) * 64 + ch] = (__bf16)hh;
    }
  } else {
    constexpr int COUT = TILES * 16;
    for (int idx = tid; idx < COUT * NPX; idx += NTHREADS) {
      const int ch = idx % COUT;
      const int n  = idx / COUT;
      float v = sOut[ch * NPX + n];
      v = v > 0.0f ? v : 0.0f;
      out[(pixBase + n) * COUT + ch] = (__bf16)v;
    }
  }
}

// ---------------------------------------------------------------- 1x1 conv (32->1) + sigmoid
__global__ void dec3_kernel(const __bf16* __restrict__ z2, const float* __restrict__ w,
                            const float* __restrict__ b, float* __restrict__ out, int total) {
  int i = blockIdx.x * blockDim.x + threadIdx.x;
  if (i >= total) return;
  float s = b[0];
  const __bf16* zp = z2 + (size_t)i * 32;
#pragma unroll
  for (int c = 0; c < 32; ++c) s += (float)zp[c] * w[c];
  out[i] = sigf(s);
}

// ---------------------------------------------------------------- launch
extern "C" void kernel_launch(void* const* d_in, const int* in_sizes, int n_in,
                              void* d_out, int out_size, void* d_ws, size_t ws_size,
                              hipStream_t stream) {
  const float*         ev     = (const float*)d_in[0];
  const unsigned char* mask   = (const unsigned char*)d_in[1];   // jnp bool = 1 byte
  const float*         emb_w  = (const float*)d_in[2];
  const float*         emb_b  = (const float*)d_in[3];
  const float*         lstm_w = (const float*)d_in[4];
  const float*         lstm_b = (const float*)d_in[5];
  const float*         dec1_w = (const float*)d_in[6];
  const float*         dec1_b = (const float*)d_in[7];
  const float*         dec2_w = (const float*)d_in[8];
  const float*         dec2_b = (const float*)d_in[9];
  const float*         dec3_w = (const float*)d_in[10];
  const float*         dec3_b = (const float*)d_in[11];

  char* ws = (char*)d_ws;
  float*  grid32 = (float*) (ws + 0);            // 64 MB  [B,H,W,128] f32
  float*  counts = (float*) (ws + 67108864);     // 512 KB [B,H,W]
  __bf16* gridbf = (__bf16*)(ws + 67633152);     // 32 MB  [B,H,W,128] bf16
  __bf16* hbf    = (__bf16*)(ws + 101187584);    // 16 MB  [B,H,W,64]
  __bf16* z1     = (__bf16*)(ws + 117964800);    // 16 MB  [B,H,W,64]
  __bf16* z2     = (__bf16*)(ws + 134742016);    //  8 MB  [B,H,W,32]
  __bf16* wp1    = (__bf16*)(ws + 143130624);    // 576 KB packed lstm weights (grid part only)
  __bf16* wp2    = (__bf16*)(ws + 143720448);    //  72 KB packed dec1 weights
  __bf16* wp3    = (__bf16*)(ws + 143794176);    //  36 KB packed dec2 weights

  // zero accumulation grid + counts (contiguous region)
  {
    const int n = BB * HH * WW * DD + BB * HH * WW;
    zero_f32_kernel<<<(n + 255) / 256, 256, 0, stream>>>(grid32, n);
  }

  // pack weights into WMMA A-fragment order (lstm: only first 128 in-channels matter; h0 == 0)
  pack_weights_kernel<192, 128, 256><<<(16 * 36 * 512 + 255) / 256, 256, 0, stream>>>(lstm_w, wp1, 16 * 36 * 512);
  pack_weights_kernel< 64,  64,  64><<<( 4 * 18 * 512 + 255) / 256, 256, 0, stream>>>(dec1_w, wp2,  4 * 18 * 512);
  pack_weights_kernel< 64,  64,  32><<<( 2 * 18 * 512 + 255) / 256, 256, 0, stream>>>(dec2_w, wp3,  2 * 18 * 512);

  // embed + scatter
  embed_scatter_kernel<<<BB * NN, DD, 0, stream>>>(ev, mask, emb_w, emb_b, grid32, counts);

  // normalize + bf16 convert
  normalize_kernel<<<(BB * HH * WW * DD + 255) / 256, 256, 0, stream>>>(grid32, counts, gridbf, BB * HH * WW * DD);

  // WMMA convs: 2048 blocks = B * H * (W/64) strips, NT=4 pixel tiles per wave
  const int nBlocks = BB * HH * (WW / 64);
  conv3x3_wmma_kernel<128, 12, 4, true ><<<nBlocks, 12 * 32, 0, stream>>>(gridbf, wp1, lstm_b, hbf);
  conv3x3_wmma_kernel< 64,  4, 4, false><<<nBlocks,  4 * 32, 0, stream>>>(hbf,    wp2, dec1_b, z1);
  conv3x3_wmma_kernel< 64,  2, 4, false><<<nBlocks,  2 * 32, 0, stream>>>(z1,     wp3, dec2_b, z2);

  // final 1x1 conv + sigmoid -> f32 output [B,H,W]
  dec3_kernel<<<(BB * HH * WW + 255) / 256, 256, 0, stream>>>(z2, dec3_w, dec3_b, (float*)d_out, BB * HH * WW);
}
